// Clustered_Attention_Chunking_89172111000174
// MI455X (gfx1250) — matrix operand order repaired and checked by reading, not compiled, not run on hardware
//
#include <hip/hip_runtime.h>
#include <hip/hip_bf16.h>

typedef __attribute__((ext_vector_type(16))) _Float16 v16h;
typedef __attribute__((ext_vector_type(8)))  _Float16 v8h;
typedef __attribute__((ext_vector_type(8)))  float    v8f;
typedef __attribute__((ext_vector_type(4)))  unsigned int v4u;
typedef __attribute__((ext_vector_type(8)))  int      v8i;
typedef __attribute__((ext_vector_type(4)))  int      v4i;

#define NSEQ 2048
#define CLEN 50
#define CPAD 64
#define EDIM 256
#define NHEAD 4
#define DHEAD 64
#define NCLUS 16
#define CHUNK 128
#define LN_EPS 1e-12f

// ---------------- WMMA helpers (CDNA5 16x16x32 f16 -> f32) ----------------

__device__ inline v8f wmma_f16(v16h a, v16h b, v8f c) {
    return __builtin_amdgcn_wmma_f32_16x16x32_f16(
        /*neg_a=*/false, a, /*neg_b=*/false, b,
        /*c_mod=*/(short)0, c, /*reuse_a=*/false, /*reuse_b=*/false);
}

// A-fragment (16x32, row-major [M][K], ld in halfs).
// ISA layout: half i of lane holds K = (i>>3)*16 + hi*8 + (i&7), M = lane&15.
__device__ inline v16h afrag(const _Float16* base, int row0, int k0, int ld) {
    int lane = threadIdx.x & 31;
    int m = lane & 15, hi = lane >> 4;
    const _Float16* p = base + (size_t)(row0 + m) * ld + k0 + hi * 8;
    v8h lo = *(const v8h*)p;
    v8h hv = *(const v8h*)(p + 16);
    return __builtin_shufflevector(lo, hv, 0,1,2,3,4,5,6,7,8,9,10,11,12,13,14,15);
}

// B-fragment (32x16) from column-major storage Bcm[col][k] (ld in halfs).
// ISA layout: half i of lane holds K = (i>>3)*16 + hi*8 + (i&7), N = lane&15.
__device__ inline v16h bfrag(const _Float16* base, int col0, int k0, int ld) {
    int lane = threadIdx.x & 31;
    int c = lane & 15, hi = lane >> 4;
    const _Float16* p = base + (size_t)(col0 + c) * ld + k0 + hi * 8;
    v8h lo = *(const v8h*)p;
    v8h hv = *(const v8h*)(p + 16);
    return __builtin_shufflevector(lo, hv, 0,1,2,3,4,5,6,7,8,9,10,11,12,13,14,15);
}

// ---------------- TDM: 2-D tile global->LDS with hardware row padding ----------------
// Moves `rows` rows of `row_units` 8-byte elements; after each full row
// (pad_interval: bytesStored/8 >= 1<<pi) the LDS address advances by
// (pad_amount+1) DWORDs. D# layout per CDNA5 ISA ch.8.
__device__ inline void tdm_load_2d(const void* gaddr, unsigned int lds_byte_off,
                                   int rows, int row_units, int pi, int pa) {
    unsigned long long ga = (unsigned long long)(uintptr_t)gaddr;
    v4u g0;
    g0[0] = 1u;                                          // count=1 (valid), user mode
    g0[1] = lds_byte_off;                                // lds_addr
    g0[2] = (unsigned int)(ga & 0xffffffffu);            // global_addr[31:0]
    g0[3] = (unsigned int)((ga >> 32) & 0x01ffffffu) | (2u << 30);  // addr[56:32] | type=2
    v8i g1;
    unsigned int w0 = (3u << 16)                         // data_size = 8B
                    | (1u << 20)                         // pad_enable
                    | ((unsigned int)pi << 22)           // pad_interval
                    | ((unsigned int)pa << 25);          // pad_amount
    g1[0] = (int)w0;
    g1[1] = (int)((unsigned int)(row_units & 0xffff) << 16);             // tensor_dim0 lo16
    g1[2] = (int)(((unsigned int)row_units >> 16) |
                  ((unsigned int)(rows & 0xffff) << 16));                // dim0 hi | dim1 lo
    g1[3] = (int)(((unsigned int)rows >> 16) |
                  ((unsigned int)(row_units & 0xffff) << 16));           // dim1 hi | tile_dim0
    g1[4] = (int)(unsigned int)(rows & 0xffff);                          // tile_dim1 | tile_dim2=0
    g1[5] = (int)(unsigned int)row_units;                                // dim0_stride lo32
    g1[6] = 0;
    g1[7] = 0;
    v4i z4 = {0, 0, 0, 0};
    v8i z8 = {0, 0, 0, 0, 0, 0, 0, 0};
    // 6-arg form on this toolchain: (g0, g1, g2, g3, extra group, cpol)
    __builtin_amdgcn_tensor_load_to_lds(g0, g1, z4, z4, z8, 0);
}

__device__ inline unsigned int lds_off(const void* p) {
    return (unsigned int)(uintptr_t)p;   // LDS aperture: low 32 bits = LDS offset
}

// ---------------- Kernel 1: stable counting sort of cluster ids ----------------

__global__ void k_sort(const int* __restrict__ cluster_id, int* __restrict__ sorted_idx) {
    __shared__ int cnt[NCLUS];
    __shared__ int off[NCLUS];
    int tid = threadIdx.x;
    if (tid < NCLUS) cnt[tid] = 0;
    __syncthreads();
    for (int i = tid; i < NSEQ; i += 256)
        atomicAdd(&cnt[cluster_id[i & 1023]], 1);   // cid = concat(cluster_id, cluster_id)
    __syncthreads();
    if (tid == 0) {
        int s = 0;
        for (int c = 0; c < NCLUS; c++) { off[c] = s; s += cnt[c]; }
    }
    __syncthreads();
    if (tid < NCLUS) {               // stable placement: one thread per cluster
        int pos = off[tid];
        for (int i = 0; i < NSEQ; i++)
            if (cluster_id[i & 1023] == tid) sorted_idx[pos++] = i;
    }
}

// ---------------- Kernel 2: convert weights f32 -> f16 ----------------

__global__ void k_wconv(const float* __restrict__ Wq, const float* __restrict__ Wk,
                        const float* __restrict__ Wv, const float* __restrict__ Wd,
                        _Float16* __restrict__ W16) {
    int m = blockIdx.y;
    int i = blockIdx.x * 256 + threadIdx.x;      // 0 .. 65535
    const float* src = (m == 0) ? Wq : (m == 1) ? Wk : (m == 2) ? Wv : Wd;
    W16[(size_t)m * EDIM * EDIM + i] = (_Float16)src[i];
}

// ---------------- Kernel 3: QKV projection (per sorted sequence) ----------------

__global__ __launch_bounds__(256)
void k_qkv(const float* __restrict__ seq, const int* __restrict__ sorted,
           const _Float16* __restrict__ W16,
           const float* __restrict__ bq, const float* __restrict__ bk,
           const float* __restrict__ bv,
           _Float16* __restrict__ Q16, _Float16* __restrict__ K16,
           _Float16* __restrict__ Vt16) {
    __shared__ _Float16 Xs[CPAD * 264];   // X (f16), rows padded to 264 halfs
    __shared__ _Float16 Ws[32 * 264];     // weight tile (TDM-loaded, hw-padded)

    int n = blockIdx.x;
    int src = sorted[n];
    int tid = threadIdx.x;
    int lane = tid & 31, wave = tid >> 5;
    int hi = lane >> 4, ln = lane & 15;
    int qt = wave >> 1, cth = wave & 1;

    for (int i = tid; i < CPAD * EDIM; i += 256) {
        int r = i >> 8, c = i & 255;
        float v = (r < CLEN) ? seq[((size_t)src * CLEN + r) * EDIM + c] : 0.0f;
        Xs[r * 264 + c] = (_Float16)v;
    }

    unsigned int wsOff = lds_off(Ws);
    for (int m = 0; m < 3; m++) {
        const _Float16* W = W16 + (size_t)m * EDIM * EDIM;
        const float* bias = (m == 0) ? bq : (m == 1) ? bk : bv;
        float qscale = (m == 0) ? (1.0f / 16.0f) : 1.0f;   // fold 1/sqrt(E) into Q
        for (int nt = 0; nt < 8; nt++) {
            __syncthreads();                 // previous tile fully consumed
            if (wave == 0) {                 // TDM: 32 rows x 512B, +16B pad per row
                tdm_load_2d(W + (size_t)nt * 32 * EDIM, wsOff, 32, 64, 6, 3);
                __builtin_amdgcn_s_wait_tensorcnt(0);
            }
            __syncthreads();
            v8f acc = {};
            #pragma unroll
            for (int k0 = 0; k0 < EDIM; k0 += 32) {
                v16h a = afrag(Xs, qt * 16, k0, 264);
                v16h b = bfrag(Ws, cth * 16, k0, 264);
                acc = wmma_f16(a, b, acc);
            }
            int col = nt * 32 + cth * 16 + ln;
            int h = col >> 6, d = col & 63;
            float bc = bias[col];
            #pragma unroll
            for (int r = 0; r < 8; r++) {
                int q = qt * 16 + hi * 8 + r;
                float v = (acc[r] + bc) * qscale;
                size_t hb = (((size_t)n * NHEAD + h) << 12);  // 64*64 per (n,h)
                if (m == 2) Vt16[hb + (size_t)d * CPAD + q] = (_Float16)v;  // V transposed
                else if (m == 1) K16[hb + (size_t)q * DHEAD + d] = (_Float16)v;
                else Q16[hb + (size_t)q * DHEAD + d] = (_Float16)v;
            }
        }
    }
}

// ---------------- Kernel 4: windowed attention (per sequence, per head) ----------------

__global__ __launch_bounds__(128)
void k_attn(const _Float16* __restrict__ Q16, const _Float16* __restrict__ K16,
            const _Float16* __restrict__ Vt16, const float* __restrict__ mask,
            _Float16* __restrict__ ctx16) {
    __shared__ _Float16 Ks[2][CPAD * 72];   // K tiles, both windows (TDM, hw-padded)
    __shared__ _Float16 Vs[2][CPAD * 72];   // V^T tiles, both windows
    __shared__ _Float16 Ps[4][16 * 72];     // per-wave probability tile

    int n = blockIdx.x >> 2, h = blockIdx.x & 3;
    int tid = threadIdx.x, lane = tid & 31, wave = tid >> 5;
    int hi = lane >> 4, ln = lane & 15;
    int qt = wave;

    int g = n >> 7, b = n & (CHUNK - 1);
    int i1 = (g > 0) ? g - 1 : 0;
    int nn[2] = { i1 * CHUNK + b, (i1 + 1) * CHUNK + b };

    const _Float16* Qp = Q16 + (((size_t)n * NHEAD + h) << 12);
    const float* mp = mask + (size_t)n * CLEN * CLEN;   // mask indexed by SORTED position

    if (wave == 0) {   // stage K/V for both windows once: 64 rows x 128B, +16B pad
        #pragma unroll
        for (int w = 0; w < 2; w++) {
            size_t hb = (((size_t)nn[w] * NHEAD + h) << 12);
            tdm_load_2d(K16 + hb, lds_off(Ks[w]), CPAD, 16, 4, 3);
            tdm_load_2d(Vt16 + hb, lds_off(Vs[w]), CPAD, 16, 4, 3);
        }
        __builtin_amdgcn_s_wait_tensorcnt(0);
    }

    v16h aQ0 = afrag(Qp, qt * 16, 0, DHEAD);
    v16h aQ1 = afrag(Qp, qt * 16, 32, DHEAD);
    __syncthreads();

    v8f o[4] = {};
    _Float16* Pw = &Ps[wave][0];

    for (int w = 0; w < 2; w++) {
        const _Float16* Kp = &Ks[w][0];
        const _Float16* Vp = &Vs[w][0];

        v8f s[4];
        #pragma unroll
        for (int ct = 0; ct < 4; ct++) {
            v8f acc = {};
            acc = wmma_f16(aQ0, bfrag(Kp, ct * 16, 0, 72), acc);
            acc = wmma_f16(aQ1, bfrag(Kp, ct * 16, 32, 72), acc);
            s[ct] = acc;
        }
        // mask add + pad-key kill
        #pragma unroll
        for (int ct = 0; ct < 4; ct++) {
            int key = ct * 16 + ln;
            #pragma unroll
            for (int r = 0; r < 8; r++) {
                int q = qt * 16 + hi * 8 + r;
                float mv = (key < CLEN) ? ((q < CLEN) ? mp[q * CLEN + key] : 0.0f) : -1e30f;
                s[ct][r] += mv;
            }
        }
        // row softmax: row lives across 16 lanes of one half-wave
        #pragma unroll
        for (int r = 0; r < 8; r++) {
            float mx = fmaxf(fmaxf(s[0][r], s[1][r]), fmaxf(s[2][r], s[3][r]));
            mx = fmaxf(mx, __shfl_xor(mx, 1));
            mx = fmaxf(mx, __shfl_xor(mx, 2));
            mx = fmaxf(mx, __shfl_xor(mx, 4));
            mx = fmaxf(mx, __shfl_xor(mx, 8));
            float p0 = __expf(s[0][r] - mx);
            float p1 = __expf(s[1][r] - mx);
            float p2 = __expf(s[2][r] - mx);
            float p3 = __expf(s[3][r] - mx);
            float sum = p0 + p1 + p2 + p3;
            sum += __shfl_xor(sum, 1);
            sum += __shfl_xor(sum, 2);
            sum += __shfl_xor(sum, 4);
            sum += __shfl_xor(sum, 8);
            float rinv = 1.0f / sum;
            int row = hi * 8 + r;
            Pw[row * 72 +  0 + ln] = (_Float16)(p0 * rinv);
            Pw[row * 72 + 16 + ln] = (_Float16)(p1 * rinv);
            Pw[row * 72 + 32 + ln] = (_Float16)(p2 * rinv);
            Pw[row * 72 + 48 + ln] = (_Float16)(p3 * rinv);
        }
        // O += P @ V  (wave-private LDS; in-wave DS ordering covers the hazard)
        v16h aP0 = afrag(Pw, 0, 0, 72);
        v16h aP1 = afrag(Pw, 0, 32, 72);
        #pragma unroll
        for (int ct = 0; ct < 4; ct++) {
            o[ct] = wmma_f16(aP0, bfrag(Vp, ct * 16, 0, 72), o[ct]);
            o[ct] = wmma_f16(aP1, bfrag(Vp, ct * 16, 32, 72), o[ct]);
        }
    }
    // store ctx = 0.5*(O1+O2), head-interleaved back into [n][c][E]
    #pragma unroll
    for (int ct = 0; ct < 4; ct++) {
        int col = h * DHEAD + ct * 16 + ln;
        #pragma unroll
        for (int r = 0; r < 8; r++) {
            int q = qt * 16 + hi * 8 + r;
            ctx16[((size_t)n * CPAD + q) * EDIM + col] = (_Float16)(o[ct][r] * 0.5f);
        }
    }
}

// ------- Kernel 5: output proj + bias + residual + LayerNorm + scatter-unsort -------

__global__ __launch_bounds__(256)
void k_out(const _Float16* __restrict__ ctx16, const _Float16* __restrict__ Wd16,
           const float* __restrict__ bd, const float* __restrict__ seq,
           const int* __restrict__ sorted, const float* __restrict__ gamma,
           const float* __restrict__ beta, float* __restrict__ out) {
    __shared__ _Float16 Cs[CPAD * 264];
    __shared__ _Float16 Ws[32 * 264];
    __shared__ float redS[4][16][2];
    __shared__ float redQ[4][16][2];

    int n = blockIdx.x;
    int src = sorted[n];
    int tid = threadIdx.x, lane = tid & 31, wave = tid >> 5;
    int hi = lane >> 4, ln = lane & 15;
    int qt = wave >> 1, wh = wave & 1;   // wave owns q-tile qt, column half wh (128 cols)

    unsigned int wsOff = lds_off(Ws);
    if (wave == 0) {   // ctx tile: 64 rows x 512B, +16B pad per row
        tdm_load_2d(ctx16 + (size_t)n * CPAD * EDIM, lds_off(Cs), CPAD, 64, 6, 3);
    }

    v8f acc[8] = {};
    for (int nt = 0; nt < 8; nt++) {
        __syncthreads();
        if (wave == 0) {
            tdm_load_2d(Wd16 + (size_t)nt * 32 * EDIM, wsOff, 32, 64, 6, 3);
            __builtin_amdgcn_s_wait_tensorcnt(0);   // also covers the Cs load (in-order)
        }
        __syncthreads();
        if ((nt >> 2) == wh) {           // wave-uniform predicate (EXEC stays all-ones)
            int f0 = (nt & 3) * 2;
            #pragma unroll
            for (int k0 = 0; k0 < EDIM; k0 += 32) {
                v16h a  = afrag(Cs, qt * 16, k0, 264);
                v16h b0 = bfrag(Ws, 0, k0, 264);
                v16h b1 = bfrag(Ws, 16, k0, 264);
                acc[f0]     = wmma_f16(a, b0, acc[f0]);
                acc[f0 + 1] = wmma_f16(a, b1, acc[f0 + 1]);
            }
        }
    }

    // x = h + bd + seq_sorted ; fragment f covers col = wh*128 + f*16 + ln
    float x[8][8];
    #pragma unroll
    for (int f = 0; f < 8; f++) {
        int col = wh * 128 + f * 16 + ln;
        float bv = bd[col];
        #pragma unroll
        for (int r = 0; r < 8; r++) {
            int q = qt * 16 + hi * 8 + r;
            float xv = acc[f][r] + bv;
            if (q < CLEN) xv += seq[((size_t)src * CLEN + q) * EDIM + col];
            x[f][r] = xv;
        }
    }
    // per-row partial stats (this wave covers 128 of 256 cols)
    #pragma unroll
    for (int r = 0; r < 8; r++) {
        float s = 0.f, sq = 0.f;
        #pragma unroll
        for (int f = 0; f < 8; f++) { s += x[f][r]; sq += x[f][r] * x[f][r]; }
        s  += __shfl_xor(s, 1);  s  += __shfl_xor(s, 2);
        s  += __shfl_xor(s, 4);  s  += __shfl_xor(s, 8);
        sq += __shfl_xor(sq, 1); sq += __shfl_xor(sq, 2);
        sq += __shfl_xor(sq, 4); sq += __shfl_xor(sq, 8);
        if (ln == 0) { redS[qt][hi * 8 + r][wh] = s; redQ[qt][hi * 8 + r][wh] = sq; }
    }
    __syncthreads();
    #pragma unroll
    for (int r = 0; r < 8; r++) {
        int row = hi * 8 + r;
        int q = qt * 16 + row;
        float s  = redS[qt][row][0] + redS[qt][row][1];
        float sq = redQ[qt][row][0] + redQ[qt][row][1];
        float mu = s * (1.0f / EDIM);
        float var = sq * (1.0f / EDIM) - mu * mu;
        float rstd = rsqrtf(var + LN_EPS);
        if (q < CLEN) {
            #pragma unroll
            for (int f = 0; f < 8; f++) {
                int col = wh * 128 + f * 16 + ln;
                float y = (x[f][r] - mu) * rstd * gamma[col] + beta[col];
                out[((size_t)src * CLEN + q) * EDIM + col] = y;   // scatter unsort
            }
        }
    }
}

// ---------------- launch ----------------

extern "C" void kernel_launch(void* const* d_in, const int* in_sizes, int n_in,
                              void* d_out, int out_size, void* d_ws, size_t ws_size,
                              hipStream_t stream) {
    (void)in_sizes; (void)n_in; (void)out_size; (void)ws_size;
    const float* seq   = (const float*)d_in[0];
    const float* mask  = (const float*)d_in[1];
    const int*   cid   = (const int*)d_in[2];
    const float* Wq    = (const float*)d_in[3];
    const float* bq    = (const float*)d_in[4];
    const float* Wk    = (const float*)d_in[5];
    const float* bk    = (const float*)d_in[6];
    const float* Wv    = (const float*)d_in[7];
    const float* bv    = (const float*)d_in[8];
    const float* Wd    = (const float*)d_in[9];
    const float* bd    = (const float*)d_in[10];
    const float* gamma = (const float*)d_in[11];
    const float* beta  = (const float*)d_in[12];
    float* out = (float*)d_out;

    char* ws = (char*)d_ws;
    int*      sorted = (int*)ws;                                  //   8 KB
    _Float16* W16    = (_Float16*)(ws + 8192);                    // 512 KB (4 x 256x256 f16)
    _Float16* Q16    = (_Float16*)(ws + 532480);                  //  64 MB
    _Float16* K16    = Q16 + (size_t)NSEQ * NHEAD * CPAD * DHEAD; //  64 MB
    _Float16* Vt16   = K16 + (size_t)NSEQ * NHEAD * CPAD * DHEAD; //  64 MB
    _Float16* CT16   = Vt16 + (size_t)NSEQ * NHEAD * CPAD * DHEAD;//  64 MB

    k_sort<<<1, 256, 0, stream>>>(cid, sorted);
    k_wconv<<<dim3(256, 4), 256, 0, stream>>>(Wq, Wk, Wv, Wd, W16);
    k_qkv<<<NSEQ, 256, 0, stream>>>(seq, sorted, W16, bq, bk, bv, Q16, K16, Vt16);
    k_attn<<<NSEQ * NHEAD, 128, 0, stream>>>(Q16, K16, Vt16, mask, CT16);
    k_out<<<NSEQ, 256, 0, stream>>>(CT16, W16 + 3 * EDIM * EDIM, bd, seq, sorted,
                                    gamma, beta, out);
}
